// SlidingWindowAttention_20005957665574
// MI455X (gfx1250) — compile-verified, matrix operand order checked by reading, not run on hardware
//
#include <hip/hip_runtime.h>

#define DHEAD  128
#define WIN    512
#define QT     128          // query rows per tile (16 per wave x 8 waves)
#define QCHUNK 8            // query tiles per block
#define KTILE  128          // keys per pass
#define NWAVE  8
#define KPAD   (DHEAD + 8)  // 136 shorts -> 272B row stride (16B mult, 4-dword bank skew)
#define VPAD   (WIN + 8)    // 520 shorts -> 1040B row stride

typedef __attribute__((ext_vector_type(16))) __bf16 v16bf;
typedef __attribute__((ext_vector_type(8)))  float  v8f;

struct U128 { unsigned int x, y, z, w; };

__device__ __forceinline__ unsigned short f2bf(float f) {
  union { float f; unsigned int u; } c; c.f = f;
  unsigned int r = c.u + 0x7FFFu + ((c.u >> 16) & 1u);   // round-to-nearest-even
  return (unsigned short)(r >> 16);
}

__device__ __forceinline__ v16bf ld_frag(const unsigned short* p0,
                                         const unsigned short* p1) {
  union { v16bf v; U128 u[2]; } f;
  f.u[0] = *reinterpret_cast<const U128*>(p0);   // K 0..7 (+8 per lane-half)
  f.u[1] = *reinterpret_cast<const U128*>(p1);   // K 16..23 (+8 per lane-half)
  return f.v;
}

__global__ __launch_bounds__(256, 1)
void swa_fwd_kernel(const float* __restrict__ Q, const float* __restrict__ K,
                    const float* __restrict__ V, float* __restrict__ O, int L) {
  // LDS-resident full window (bf16): K [key][d], V transposed [d][key]
  __shared__ __align__(16) unsigned short sK [WIN][KPAD];        // 139,264 B
  __shared__ __align__(16) unsigned short sVt[DHEAD][VPAD];      // 133,120 B
  // wave-private overlay: Q staging, then P (probs) each key pass
  __shared__ __align__(16) unsigned short sS [NWAVE][16][KPAD];  //  34,816 B

  const int tid  = threadIdx.x;
  const int wave = tid >> 5;
  const int lane = tid & 31;
  const int half = lane >> 4;
  const int ln   = lane & 15;

  const int bh = blockIdx.y;
  const int q0 = blockIdx.x * (QT * QCHUNK);

  const float* qp = Q + (size_t)bh * L * DHEAD;
  const float* kp = K + ((size_t)bh * L + (size_t)(L - WIN)) * DHEAD;
  const float* vp = V + ((size_t)bh * L + (size_t)(L - WIN)) * DHEAD;
  float*       op = O + (size_t)bh * L * DHEAD;

  // ---- one-time cooperative K/V window load + bf16 convert (+ V transpose) ----
  for (int i = tid; i < WIN * DHEAD / 4; i += 256) {
    const int r = (i * 4) / DHEAD, c = (i * 4) % DHEAD;
    const size_t gro = (size_t)r * DHEAD + c;
    const float4 fk = *reinterpret_cast<const float4*>(kp + gro);
    sK[r][c + 0] = f2bf(fk.x); sK[r][c + 1] = f2bf(fk.y);
    sK[r][c + 2] = f2bf(fk.z); sK[r][c + 3] = f2bf(fk.w);
    const float4 fv = *reinterpret_cast<const float4*>(vp + gro);
    sVt[c + 0][r] = f2bf(fv.x); sVt[c + 1][r] = f2bf(fv.y);
    sVt[c + 2][r] = f2bf(fv.z); sVt[c + 3][r] = f2bf(fv.w);
  }
  __syncthreads();   // K/V resident for block lifetime; no barriers after this

  const float scale = 0.08838834764831845f;  // 1/sqrt(128), folded into Q

  for (int qi = 0; qi < QCHUNK; ++qi) {
    const int qbase = q0 + qi * QT;

    // ---- stage this wave's 16 Q rows -> LDS bf16, pre-scaled (wave-private) ----
    {
      const float* qw = qp + (size_t)(qbase + wave * 16) * DHEAD;
      for (int i = lane; i < 16 * DHEAD / 4; i += 32) {
        const int r = (i * 4) / DHEAD, c = (i * 4) % DHEAD;
        const float4 f = *reinterpret_cast<const float4*>(qw + (size_t)r * DHEAD + c);
        sS[wave][r][c + 0] = f2bf(f.x * scale); sS[wave][r][c + 1] = f2bf(f.y * scale);
        sS[wave][r][c + 2] = f2bf(f.z * scale); sS[wave][r][c + 3] = f2bf(f.w * scale);
      }
      if (qi + 1 < QCHUNK) {  // prefetch next Q tile, 128B granules
        const float* qn = qp + (size_t)(qbase + QT + wave * 16) * DHEAD;
        for (int i = lane; i < 64; i += 32)
          __builtin_prefetch(qn + i * 32, 0, 1);
      }
    }

    // ---- Q A-fragments into registers (frees overlay for P) ----
    v16bf aQ[4];
#pragma unroll
    for (int c = 0; c < 4; ++c)
      aQ[c] = ld_frag(&sS[wave][ln][c * 32 + half * 8],
                      &sS[wave][ln][c * 32 + 16 + half * 8]);

    v8f   o[8];
    float lrow[8];
#pragma unroll
    for (int i = 0; i < 8; ++i) {
#pragma unroll
      for (int j = 0; j < 8; ++j) o[i][j] = 0.0f;
      lrow[i] = 0.0f;
    }

    const int rowbase = qbase + wave * 16 + half * 8;  // + r gives row l
    const int qmaxr   = qbase + QT - 1;
    const int nkt     = (qmaxr / KTILE + 1) < (WIN / KTILE) ? (qmaxr / KTILE + 1)
                                                            : (WIN / KTILE);

    for (int kt = 0; kt < nkt; ++kt) {
      const int  kb       = kt * KTILE;
      const bool may_mask = (kb + KTILE - 1) > rowbase;

      // ---- S halves: 4 key sub-tiles at a time keeps only s[4] live ----
#pragma unroll
      for (int h = 0; h < 2; ++h) {
        v8f s[4];
#pragma unroll
        for (int n4 = 0; n4 < 4; ++n4) {
          const int nst = h * 4 + n4;
          v8f acc;
#pragma unroll
          for (int j = 0; j < 8; ++j) acc[j] = 0.0f;
#pragma unroll
          for (int c = 0; c < 4; ++c) {
            const v16bf b = ld_frag(&sK[kb + nst * 16 + ln][c * 32 + half * 8],
                                    &sK[kb + nst * 16 + ln][c * 32 + 16 + half * 8]);
            acc = __builtin_amdgcn_wmma_f32_16x16x32_bf16(
                false, aQ[c], false, b, (short)0, acc, false, false);
          }
          s[n4] = acc;
        }

        // ---- unnormalized probs: p = exp(s) (scale pre-folded); mask -> 0 ----
#pragma unroll
        for (int n4 = 0; n4 < 4; ++n4) {
          const int nst = h * 4 + n4;
          const int j   = kb + nst * 16 + ln;
#pragma unroll
          for (int r = 0; r < 8; ++r) {
            float p = __expf(s[n4][r]);
            if (may_mask && j > rowbase + r) p = 0.0f;
            lrow[r] += p;                                  // per-lane partial row sum
            sS[wave][r + half * 8][nst * 16 + ln] = f2bf(p);
          }
        }
      }

      // ---- O += P.V : 4 key chunks of 32 x 8 d-tiles ----
#pragma unroll
      for (int c = 0; c < 4; ++c) {
        const v16bf aP = ld_frag(&sS[wave][ln][c * 32 + half * 8],
                                 &sS[wave][ln][c * 32 + 16 + half * 8]);
#pragma unroll
        for (int dt = 0; dt < 8; ++dt) {
          const v16bf bV = ld_frag(&sVt[dt * 16 + ln][kb + c * 32 + half * 8],
                                   &sVt[dt * 16 + ln][kb + c * 32 + 16 + half * 8]);
          o[dt] = __builtin_amdgcn_wmma_f32_16x16x32_bf16(
              false, aP, false, bV, (short)0, o[dt], false, false);
        }
      }
    } // kt

    // ---- epilogue: one cross-lane row-sum reduction, then normalize+store ----
#pragma unroll
    for (int r = 0; r < 8; ++r) {
#pragma unroll
      for (int off = 1; off < 16; off <<= 1)
        lrow[r] += __shfl_xor(lrow[r], off, 32);
      lrow[r] = 1.0f / lrow[r];
    }
#pragma unroll
    for (int dt = 0; dt < 8; ++dt) {
#pragma unroll
      for (int r = 0; r < 8; ++r) {
        const int row = qbase + wave * 16 + half * 8 + r;
        op[(size_t)row * DHEAD + dt * 16 + ln] = o[dt][r] * lrow[r];
      }
    }
  } // qi
}

extern "C" void kernel_launch(void* const* d_in, const int* in_sizes, int n_in,
                              void* d_out, int out_size, void* d_ws, size_t ws_size,
                              hipStream_t stream) {
  (void)n_in; (void)out_size; (void)d_ws; (void)ws_size;
  const float* q = (const float*)d_in[0];
  const float* k = (const float*)d_in[1];
  const float* v = (const float*)d_in[2];
  float* out = (float*)d_out;

  const int L  = 4096;                       // per reference setup
  const int BH = in_sizes[0] / (L * DHEAD);  // B*H = 128
  dim3 grid(L / (QT * QCHUNK), BH);          // (4, 128)
  swa_fwd_kernel<<<grid, 256, 0, stream>>>(q, k, v, out, L);
}
